// minGRULayer_58488864637198
// MI455X (gfx1250) — matrix operand order repaired and checked by reading, not compiled, not run on hardware
//
#include <hip/hip_runtime.h>
#include <math.h>

// ---------------------------------------------------------------------------
// minGRU bidirectional layer, B=8 D=512 L=8192 (f32 in/out), MI455X / gfx1250.
//  pass 0a: x[b][d][l] f32 -> XhT[b][l][d] f16  (tiled LDS transpose, done ONCE;
//           64MB result is L2-resident on MI455X's 192MB L2, so all GEMM B-operand
//           re-reads are L2 gathers, not HBM traffic)
//  pass 0b: w_fore|w_back f32 -> f16 (straight copy, 1MB, trivially L2-resident)
//  pass 1 : LDS-free WMMA GEMM (f16 A/B, f32 acc) with fused minGRU activation
//           epilogue -> per-sequence contiguous (a,b) blocks [seq][2][8192] f32
//  pass 2 : per-sequence parallel scan; sequence staged into LDS by ONE
//           Tensor Data Mover tensor_load_to_lds (TENSORcnt), back dir reversed
// ---------------------------------------------------------------------------

typedef __attribute__((ext_vector_type(16))) _Float16 v16h;
typedef __attribute__((ext_vector_type(8)))  float    v8f;
typedef __attribute__((ext_vector_type(4)))  unsigned int v4u;
typedef __attribute__((ext_vector_type(4)))  int      v4i;
typedef __attribute__((ext_vector_type(8)))  int      v8i;
typedef __attribute__((ext_vector_type(4)))  float    v4f;

union U16 { v16h v; v4u q[2]; };

#define BB 8
#define DD 512
#define LL 8192
#define JHALF 256          // D/2 channels per direction

// ---------------- pass 0a: convert + transpose x ---------------------------
// 64(l) x 64(d) tile per block via LDS; reads coalesced in l, writes in d.
__global__ __launch_bounds__(256) void mingru_xpose_f16(const float* __restrict__ x,
                                                        _Float16* __restrict__ XhT) {
  __shared__ _Float16 tile[64 * 65];      // pad 65 to dodge bank conflicts
  const int t  = threadIdx.x;
  const int l0 = blockIdx.x * 64;
  const int d0 = blockIdx.y * 64;
  const int b  = blockIdx.z;
  const float*    xb = x   + (size_t)b * DD * LL;
  _Float16*       ob = XhT + (size_t)b * LL * DD;
  for (int it = 0; it < 16; ++it) {
    int idx = it * 256 + t;
    int i = idx >> 6;                     // d within tile
    int j = idx & 63;                     // l within tile (coalesced)
    tile[i * 65 + j] = (_Float16)xb[(size_t)(d0 + i) * LL + (l0 + j)];
  }
  __syncthreads();
  for (int it = 0; it < 16; ++it) {
    int idx = it * 256 + t;
    int j = idx >> 6;                     // l within tile
    int i = idx & 63;                     // d within tile (coalesced)
    ob[(size_t)(l0 + j) * DD + (d0 + i)] = tile[i * 65 + j];
  }
}

// ---------------- pass 0b: f32 -> f16 straight (x4) ------------------------
__global__ __launch_bounds__(256) void mingru_cvt_f16(const float* __restrict__ src,
                                                      _Float16* __restrict__ dst, int n4) {
  int i = blockIdx.x * 256 + threadIdx.x;
  if (i < n4) {
    v4f v = reinterpret_cast<const v4f*>(src)[i];
    union { _Float16 h[4]; unsigned long long u; } o;
    o.h[0] = (_Float16)v.x; o.h[1] = (_Float16)v.y;
    o.h[2] = (_Float16)v.z; o.h[3] = (_Float16)v.w;
    reinterpret_cast<unsigned long long*>(dst)[i] = o.u;
  }
}

// ---------------- pass 1: LDS-free WMMA GEMM + activation ------------------
// grid (L/128, 256/16, B*2), block 256 = 8 waves. Each block: h-rows [j0,j0+16)
// and the paired gate rows; each wave owns a 16-column subtile.
// A (16x32 f16 layout): lanes<16 hold K {0..7,16..23}, lanes>=16 {8..15,24..31},
//   row = j0 + (lane&15)  -> two b128 from Wh row (contiguous).
// B (32x16 f16 layout): lane c holds column c&15, same K order as A (lanes<16 {0..7,16..23}, lanes>=16 {8..15,24..31})
//   -> two b128 from XhT row (d contiguous per lane) — no LDS, no barriers.
__global__ __launch_bounds__(256) void mingru_gemm_act(
    const _Float16* __restrict__ Wh, const _Float16* __restrict__ XhT,
    const float* __restrict__ b_fore, const float* __restrict__ b_back,
    float* __restrict__ AB_ws) {
  const int tid  = threadIdx.x;
  const int wave = tid >> 5;
  const int lane = tid & 31;
  const int cl   = lane & 15;
  const int hi   = lane >> 4;

  const int n0  = blockIdx.x * 128;
  const int j0  = blockIdx.y * 16;
  const int z   = blockIdx.z;
  const int b   = z & 7;
  const int dir = z >> 3;

  const float* bias = dir ? b_back : b_fore;
  const int l = n0 + wave * 16 + cl;                       // this lane's column
  const _Float16* rowB  = XhT + ((size_t)b * LL + l) * DD;
  const _Float16* WrowH = Wh + (((size_t)(dir * 2 + 0) * JHALF) + j0 + cl) * DD;
  const _Float16* WrowG = Wh + (((size_t)(dir * 2 + 1) * JHALF) + j0 + cl) * DD;

  v8f accH = {0.f,0.f,0.f,0.f,0.f,0.f,0.f,0.f};
  v8f accG = {0.f,0.f,0.f,0.f,0.f,0.f,0.f,0.f};

#pragma unroll
  for (int kt = 0; kt < 16; ++kt) {
    const int k0   = kt * 32;
    const int off0 = k0 + (hi ? 8  : 0);
    const int off1 = k0 + (hi ? 24 : 16);
    const int klo  = k0 + hi * 16;
    U16 aH, aG, bm;
    aH.q[0] = *reinterpret_cast<const v4u*>(WrowH + off0);
    aH.q[1] = *reinterpret_cast<const v4u*>(WrowH + off1);
    aG.q[0] = *reinterpret_cast<const v4u*>(WrowG + off0);
    aG.q[1] = *reinterpret_cast<const v4u*>(WrowG + off1);
    bm.q[0] = *reinterpret_cast<const v4u*>(rowB + off0);
    bm.q[1] = *reinterpret_cast<const v4u*>(rowB + off1);
    if (kt < 14) __builtin_prefetch(rowB + k0 + 64, 0, 0);  // global_prefetch_b8

    accH = __builtin_amdgcn_wmma_f32_16x16x32_f16(false, aH.v, false, bm.v,
                                                  (short)0, accH, false, false);
    accG = __builtin_amdgcn_wmma_f32_16x16x32_f16(false, aG.v, false, bm.v,
                                                  (short)0, accG, false, false);
  }

  // epilogue: C layout lane<16 -> (M=r, N=lane), lane>=16 -> (M=r+8, N=lane-16)
  for (int r = 0; r < 8; ++r) {
    int j = j0 + r + hi * 8;
    float hval = accH[r] + bias[j];
    float gval = accG[r] + bias[JHALF + j];
    float aval = 1.0f / (1.0f + __expf(gval));    // sigmoid(-g) = exp(-softplus(g))
    float sg   = 1.0f / (1.0f + __expf(-gval));   // sigmoid(g)
    float gf   = (hval >= 0.0f) ? (hval + 1.0f) : __expf(hval);   // g(h)
    size_t seq = ((size_t)(dir * 8 + b) * JHALF) + j;
    size_t o   = seq * (2 * LL) + l;              // [seq][0][l] = a, [seq][1][l] = b
    AB_ws[o]      = aval;
    AB_ws[o + LL] = sg * gf;
  }
}

// ---------------- pass 2: per-sequence parallel scan -----------------------
// grid 4096 blocks (== seq index), 256 threads; (a,b) staged to LDS by one TDM
// tensor_load_to_lds (tile 8192 x 2 rows, 4B elems), then affine block scan.
__global__ __launch_bounds__(256) void mingru_scan(const float* __restrict__ AB_ws,
                                                   float* __restrict__ out) {
  extern __shared__ float sm[];
  float* aArr = sm;                 // [8192]
  float* bArr = sm + LL;            // [8192]
  float* sA   = sm + 2 * LL;        // [256]
  float* sB   = sm + 2 * LL + 256;  // [256]

  const int t   = threadIdx.x;
  const int blk = blockIdx.x;                // == seq = (dir*8+b)*256 + j
  const int j   = blk & 255;
  const int b   = (blk >> 8) & 7;
  const int dir = blk >> 11;
  const bool rev = (dir == 1);
  const size_t base = (size_t)blk * (2 * LL);

#if __has_builtin(__builtin_amdgcn_tensor_load_to_lds)
  if (t < 32) {                              // wave 0 issues the DMA (uniform branch)
    unsigned long long ga = (unsigned long long)(AB_ws + base);
    unsigned lds_off = (unsigned)(unsigned long long)aArr;   // flat LDS addr low 32b
    v4u g0;
    g0[0] = 1u;                                              // count=1, user D#
    g0[1] = lds_off;                                         // lds_addr
    g0[2] = (unsigned)(ga & 0xffffffffull);                  // global_addr lo
    g0[3] = (unsigned)((ga >> 32) & 0x01ffffffull) | (2u << 30); // addr hi | type=2
    v8i g1;
    g1[0] = (int)(2u << 16);                  // wg_mask=0, data_size=2 -> 4B elems
    g1[1] = (int)((LL & 0xffffu) << 16);      // tensor_dim0 lo16 (8192)
    g1[2] = (int)(((LL >> 16) & 0xffffu) | (2u << 16)); // dim0 hi | tensor_dim1=2
    g1[3] = (int)((LL & 0xffffu) << 16);      // dim1 hi=0 | tile_dim0=8192
    g1[4] = 2;                                // tile_dim1=2, tile_dim2=0
    g1[5] = LL;                               // tensor_dim0_stride lo32 = 8192
    g1[6] = 0;                                // stride hi | dim1_stride lo
    g1[7] = 0;                                // dim1_stride hi
    v4i z4 = {0, 0, 0, 0};
#if defined(__clang_major__) && (__clang_major__ >= 23)
    v8i z8 = {0, 0, 0, 0, 0, 0, 0, 0};
    __builtin_amdgcn_tensor_load_to_lds(g0, g1, z4, z4, z8, 0);
#else
    __builtin_amdgcn_tensor_load_to_lds(g0, g1, z4, z4, 0);
#endif
    __builtin_amdgcn_s_wait_tensorcnt(0);
  }
  __syncthreads();
#else
  for (int it = 0; it < 32; ++it) {
    int idx = it * 256 + t;
    aArr[idx] = AB_ws[base + idx];
    bArr[idx] = AB_ws[base + LL + idx];
  }
  __syncthreads();
#endif

  // local 32-step affine composition: h_out = A*h_in + Bc
  float A = 1.0f, Bc = 0.0f;
  const int s0 = t * 32;
  for (int i = 0; i < 32; ++i) {
    int s = s0 + i;
    int p = rev ? (LL - 1 - s) : s;
    float a = aArr[p], bv = bArr[p];
    A = a * A;
    Bc = a * Bc + bv;
  }
  sA[t] = A; sB[t] = Bc;
  __syncthreads();

  // Hillis-Steele inclusive scan of affine pairs across 256 threads
  for (int off = 1; off < 256; off <<= 1) {
    float pA = 0.f, pB = 0.f;
    bool v = (t >= off);
    if (v) { pA = sA[t - off]; pB = sB[t - off]; }
    __syncthreads();
    if (v) { float mA = sA[t], mB = sB[t]; sA[t] = mA * pA; sB[t] = mA * pB + mB; }
    __syncthreads();
  }

  float h = (t == 0) ? 0.0f : sB[t - 1];     // exclusive prefix applied to h0=0
  for (int i = 0; i < 32; ++i) {
    int s = s0 + i;
    int p = rev ? (LL - 1 - s) : s;
    h = aArr[p] * h + bArr[p];
    aArr[p] = h;                              // own-slot rewrite only
  }
  __syncthreads();

  const int c = dir ? (JHALF + j) : j;        // fore -> [0,256), back -> [256,512)
  const size_t obase = (((size_t)b * DD) + c) * LL;
  for (int it = 0; it < 32; ++it) {
    int idx = it * 256 + t;
    out[obase + idx] = aArr[idx];
  }
}

// ---------------------------------------------------------------------------
extern "C" void kernel_launch(void* const* d_in, const int* in_sizes, int n_in,
                              void* d_out, int out_size, void* d_ws, size_t ws_size,
                              hipStream_t stream) {
  const float* x      = (const float*)d_in[0];
  const float* w_fore = (const float*)d_in[1];
  const float* b_fore = (const float*)d_in[2];
  const float* w_back = (const float*)d_in[3];
  const float* b_back = (const float*)d_in[4];
  float* out = (float*)d_out;
  char* ws = (char*)d_ws;

  // workspace layout
  const size_t XT_OFF = 0;                                   // XhT: 64MB f16
  const size_t WH_OFF = XT_OFF + (size_t)BB * DD * LL * 2;   // Wh : 1MB f16
  const size_t AB_OFF = WH_OFF + (size_t)2 * DD * DD * 2;    // AB : 256MB f32

  _Float16* XhT = (_Float16*)(ws + XT_OFF);
  _Float16* Wh  = (_Float16*)(ws + WH_OFF);
  float*    AB  = (float*)(ws + AB_OFF);

  // pass 0: transpose+convert x; convert weights (packed [dir][512][512])
  {
    dim3 g(LL / 64, DD / 64, BB);
    mingru_xpose_f16<<<g, 256, 0, stream>>>(x, XhT);
    int n4w = (DD * DD) / 4;
    mingru_cvt_f16<<<(n4w + 255) / 256, 256, 0, stream>>>(w_fore, Wh, n4w);
    mingru_cvt_f16<<<(n4w + 255) / 256, 256, 0, stream>>>(w_back, Wh + DD * DD, n4w);
  }

  // pass 1: WMMA GEMM + fused activation
  {
    dim3 grid(LL / 128, JHALF / 16, BB * 2);
    mingru_gemm_act<<<grid, 256, 0, stream>>>(Wh, XhT, b_fore, b_back, AB);
  }

  // pass 2: scans (fore forward, back reversed), TDM-staged LDS
  {
    size_t shmem = (size_t)(2 * LL + 512) * sizeof(float);   // 66 KB
    mingru_scan<<<2 * BB * JHALF, 256, shmem, stream>>>(AB, out);
  }
}
